// Model_34411277976466
// MI455X (gfx1250) — matrix operand order repaired and checked by reading, not compiled, not run on hardware
//
#include <hip/hip_runtime.h>

typedef __attribute__((ext_vector_type(16))) _Float16 v16h;
typedef __attribute__((ext_vector_type(8)))  float    v8f;

#define HDIM 64

__device__ __forceinline__ void atomic_fadd(float* p, float v) {
  __hip_atomic_fetch_add(p, v, __ATOMIC_RELAXED, __HIP_MEMORY_SCOPE_AGENT);
}

// ---------------- zero ----------------
__global__ void zero_kernel(float* __restrict__ p, long n) {
  long i = (long)blockIdx.x * blockDim.x + threadIdx.x;
  if (i < n) p[i] = 0.0f;
}

// ---------------- degrees ----------------
__global__ void degree_kernel(const int* __restrict__ src, const int* __restrict__ dst,
                              float* __restrict__ deg_u, float* __restrict__ deg_m, int E) {
  int e = blockIdx.x * blockDim.x + threadIdx.x;
  if (e >= E) return;
  atomic_fadd(&deg_m[dst[e]], 1.0f);
  atomic_fadd(&deg_u[src[e]], 1.0f);
}

// ---------------- split weights once: w -> (hi = f16(w), lo = f16(w - hi)) ----------------
__global__ void wsplit_kernel(const float* __restrict__ w,
                              _Float16* __restrict__ hi, _Float16* __restrict__ lo, int n) {
  int i = blockIdx.x * blockDim.x + threadIdx.x;
  if (i >= n) return;
  float x = w[i];
  _Float16 h = (_Float16)x;
  hi[i] = h;
  lo[i] = (_Float16)(x - (float)h);
}

// ---------------- movie feature init: xm = movie_x @ W^T + b + movie_emb ----------------
__global__ void movie_init_kernel(const float* __restrict__ movie_x,  // [M,20]
                                  const float* __restrict__ w,        // [64,20]
                                  const float* __restrict__ b,        // [64]
                                  const float* __restrict__ movie_emb,// [M,64]
                                  float* __restrict__ xm, int M) {
  int id = blockIdx.x * blockDim.x + threadIdx.x;
  if (id >= M * HDIM) return;
  int m = id >> 6, o = id & 63;
  float acc = b[o] + movie_emb[id];
  const float* xr = movie_x + (long)m * 20;
  const float* wr = w + o * 20;
#pragma unroll
  for (int f = 0; f < 20; ++f) acc += xr[f] * wr[f];
  xm[id] = acc;
}

// ---------------- edge scatter: agg_m[dst] += x_u[src]; agg_u[src] += x_m[dst] ----------------
// one wave per edge; each lane handles 2 consecutive floats of the 64-float rows
__global__ void scatter_kernel(const float* __restrict__ xu, const float* __restrict__ xm,
                               const int* __restrict__ src, const int* __restrict__ dst,
                               float* __restrict__ agg_m, float* __restrict__ agg_u, int E) {
  int t = blockIdx.x * blockDim.x + threadIdx.x;
  int e = t >> 5, lane = t & 31;
  if (e >= E) return;
  int s = src[e], d = dst[e];
  float2 vu = ((const float2*)(xu + (long)s * HDIM))[lane];
  float2 vm = ((const float2*)(xm + (long)d * HDIM))[lane];
  float* am = agg_m + (long)d * HDIM + lane * 2;
  float* au = agg_u + (long)s * HDIM + lane * 2;
  atomic_fadd(am,     vu.x);
  atomic_fadd(am + 1, vu.y);
  atomic_fadd(au,     vm.x);
  atomic_fadd(au + 1, vm.y);
}

// ---------------- A fragment loader (split-f16 on the fly) ----------------
// A fragment (16x32 f16, MxK): lane holds row M=(lane&15); element e maps to
// K = ks*32 + 16*(lane>>4) + e  -> one contiguous 16-float run.
__device__ __forceinline__ void load_a_split(const float* __restrict__ row, int ks, int g,
                                             float scale, v16h& hi, v16h& lo) {
  int base0 = ks * 32 + 16 * g;             // multiple of 8 -> float4 aligned
  const float4* p = (const float4*)row;
  float4 r0 = p[(base0 >> 2) + 0];
  float4 r1 = p[(base0 >> 2) + 1];
  float4 r2 = p[(base0 >> 2) + 2];         // base0 + 8
  float4 r3 = p[(base0 >> 2) + 3];
  float v[16] = {r0.x, r0.y, r0.z, r0.w, r1.x, r1.y, r1.z, r1.w,
                 r2.x, r2.y, r2.z, r2.w, r3.x, r3.y, r3.z, r3.w};
#pragma unroll
  for (int i = 0; i < 16; ++i) {
    float x = v[i] * scale;
    _Float16 h = (_Float16)x;
    hi[i] = h;
    lo[i] = (_Float16)(x - (float)h);
  }
}

// B fragment (32x16 f16, KxN) for Y = X * W^T: B[k,n] = W[n,k]; with the
// pre-split row-major f16 weight array this is a single aligned 32B v16h load:
// elements are w16[col*64 + ks*32 + 16*(lane>>4) + e], e = 0..15.
__device__ __forceinline__ v16h load_b16(const _Float16* __restrict__ w16, int col, int ks, int g) {
  return *(const v16h*)(w16 + (size_t)col * HDIM + ks * 32 + 16 * g);
}

#define WMMA(A, B, C) \
  __builtin_amdgcn_wmma_f32_16x16x32_f16(false, (A), false, (B), (short)0, (C), false, false)

// ---------------- fused SAGE transform: out = relu?( (agg/deg)@wl^T + bl + xin@wr^T ) ----------------
// one wave32 per 16-row node tile; split-f16 WMMA with f32 accumulation.
// Loop order: k-step outer (A fragments transient), col-tile inner (4 persistent accumulators)
// to keep register pressure low.
__global__ void sage_gemm_kernel(const float* __restrict__ agg, const float* __restrict__ deg,
                                 const float* __restrict__ xin, float* __restrict__ xout,
                                 const _Float16* __restrict__ wlhi, const _Float16* __restrict__ wllo,
                                 const float* __restrict__ bvec,
                                 const _Float16* __restrict__ wrhi, const _Float16* __restrict__ wrlo,
                                 int N, int do_relu) {
  int tile = blockIdx.x * (blockDim.x >> 5) + (threadIdx.x >> 5);  // wave-uniform
  if (tile * 16 >= N) return;                                     // whole-wave exit -> EXEC all-1s otherwise
  int lane = threadIdx.x & 31;
  int g = lane >> 4;
  int r = tile * 16 + (lane & 15);
  if (r >= N) r = N - 1;  // safety (N is a multiple of 16 here)
  float invd = 1.0f / fmaxf(deg[r], 1.0f);
  const float* aggrow = agg + (long)r * HDIM;
  const float* xrow   = xin + (long)r * HDIM;

  int n = lane & 15;
  v8f acc[4] = {};   // one 16x16 f32 accumulator per column tile

#pragma unroll
  for (int ks = 0; ks < 2; ++ks) {
    // A fragments for this k-step only (4 x v16h live)
    v16h aMh, aMl, aXh, aXl;
    load_a_split(aggrow, ks, g, invd, aMh, aMl);
    load_a_split(xrow,   ks, g, 1.0f, aXh, aXl);

#pragma unroll
    for (int ct = 0; ct < 4; ++ct) {
      int col = ct * 16 + n;
      // mean @ wl^T  (split product: Ah*Bh + Ah*Bl + Al*Bh)
      {
        v16h bh = load_b16(wlhi, col, ks, g);
        v16h bl = load_b16(wllo, col, ks, g);
        acc[ct] = WMMA(aMh, bh, acc[ct]);
        acc[ct] = WMMA(aMh, bl, acc[ct]);
        acc[ct] = WMMA(aMl, bh, acc[ct]);
      }
      // x_dst @ wr^T
      {
        v16h bh = load_b16(wrhi, col, ks, g);
        v16h bl = load_b16(wrlo, col, ks, g);
        acc[ct] = WMMA(aXh, bh, acc[ct]);
        acc[ct] = WMMA(aXh, bl, acc[ct]);
        acc[ct] = WMMA(aXl, bh, acc[ct]);
      }
    }
  }

#pragma unroll
  for (int ct = 0; ct < 4; ++ct) {
    int col = ct * 16 + n;
    float bias = bvec[col];
#pragma unroll
    for (int j = 0; j < 8; ++j) {
      int orow = tile * 16 + j + 8 * g;   // D layout: VGPR j -> M = j + 8*(lane>>4)
      float v = acc[ct][j] + bias;
      if (do_relu) v = fmaxf(v, 0.0f);
      if (orow < N) xout[(long)orow * HDIM + col] = v;
    }
  }
}

// ---------------- link classifier: out[l] = dot(x_u[ls[l]], x_m[ld[l]]) ----------------
__global__ void classifier_kernel(const float* __restrict__ xu, const float* __restrict__ xm,
                                  const int* __restrict__ ls, const int* __restrict__ ld,
                                  float* __restrict__ out, int L) {
  int t = blockIdx.x * blockDim.x + threadIdx.x;
  int l = t >> 5, lane = t & 31;
  if (l >= L) return;
  int s = ls[l], d = ld[l];
  float2 a = ((const float2*)(xu + (long)s * HDIM))[lane];
  float2 b = ((const float2*)(xm + (long)d * HDIM))[lane];
  float p = a.x * b.x + a.y * b.y;
#pragma unroll
  for (int off = 16; off > 0; off >>= 1) p += __shfl_xor(p, off, 32);
  if (lane == 0) out[l] = p;
}

extern "C" void kernel_launch(void* const* d_in, const int* in_sizes, int n_in,
                              void* d_out, int out_size, void* d_ws, size_t ws_size,
                              hipStream_t stream) {
  const float* movie_x     = (const float*)d_in[0];
  const float* user_emb    = (const float*)d_in[1];
  const float* movie_emb   = (const float*)d_in[2];
  const float* movie_lin_w = (const float*)d_in[3];
  const float* movie_lin_b = (const float*)d_in[4];
  const float* lin_l_w     = (const float*)d_in[5];
  const float* lin_l_b     = (const float*)d_in[6];
  const float* lin_r_w     = (const float*)d_in[7];
  const int*   edge_src    = (const int*)d_in[8];
  const int*   edge_dst    = (const int*)d_in[9];
  const int*   label_src   = (const int*)d_in[10];
  const int*   label_dst   = (const int*)d_in[11];
  float* out = (float*)d_out;

  const int U = in_sizes[1] / HDIM;
  const int M = in_sizes[2] / HDIM;
  const int E = in_sizes[8];
  const int L = in_sizes[10];
  const int NW = 2 * 2 * HDIM * HDIM;  // elements in lin_l_w / lin_r_w (16384)

  float* ws    = (float*)d_ws;
  float* xm    = ws; ws += (size_t)M * HDIM;
  float* xu    = ws; ws += (size_t)U * HDIM;
  float* agg_m = ws; ws += (size_t)M * HDIM;   // agg_m, agg_u contiguous (zeroed together)
  float* agg_u = ws; ws += (size_t)U * HDIM;
  float* deg_m = ws; ws += M;                   // deg_m, deg_u contiguous
  float* deg_u = ws; ws += U;
  // split-f16 weight copies (32B-aligned: all preceding sizes are multiples of 32B)
  _Float16* wl_hi = (_Float16*)ws;
  _Float16* wl_lo = wl_hi + NW;
  _Float16* wr_hi = wl_lo + NW;
  _Float16* wr_lo = wr_hi + NW;

  const int B = 256;

  // degrees (graph-only, but recomputed each call for determinism/statelessness)
  long ndeg = (long)M + U;
  zero_kernel<<<(int)((ndeg + B - 1) / B), B, 0, stream>>>(deg_m, ndeg);
  degree_kernel<<<(E + B - 1) / B, B, 0, stream>>>(edge_src, edge_dst, deg_u, deg_m, E);

  // split weights (tiny, stays in L2)
  wsplit_kernel<<<(NW + B - 1) / B, B, 0, stream>>>(lin_l_w, wl_hi, wl_lo, NW);
  wsplit_kernel<<<(NW + B - 1) / B, B, 0, stream>>>(lin_r_w, wr_hi, wr_lo, NW);

  // x_m init
  movie_init_kernel<<<(M * HDIM + B - 1) / B, B, 0, stream>>>(
      movie_x, movie_lin_w, movie_lin_b, movie_emb, xm, M);

  for (int layer = 0; layer < 2; ++layer) {
    long nagg = (long)(M + U) * HDIM;
    zero_kernel<<<(int)((nagg + B - 1) / B), B, 0, stream>>>(agg_m, nagg);

    const float* xu_cur = (layer == 0) ? user_emb : xu;
    scatter_kernel<<<(E * 32 + B - 1) / B, B, 0, stream>>>(
        xu_cur, xm, edge_src, edge_dst, agg_m, agg_u, E);

    size_t off_m = (size_t)(layer * 2 + 0) * HDIM * HDIM;
    size_t off_u = (size_t)(layer * 2 + 1) * HDIM * HDIM;
    const float* bl_m = lin_l_b + (size_t)(layer * 2 + 0) * HDIM;
    const float* bl_u = lin_l_b + (size_t)(layer * 2 + 1) * HDIM;
    int relu = (layer == 0) ? 1 : 0;

    int tiles_m = (M + 15) / 16;
    int tiles_u = (U + 15) / 16;
    // 128 threads = 4 wave32 = 4 tiles per block
    sage_gemm_kernel<<<(tiles_m + 3) / 4, 128, 0, stream>>>(
        agg_m, deg_m, xm, xm,
        wl_hi + off_m, wl_lo + off_m, bl_m, wr_hi + off_m, wr_lo + off_m, M, relu);
    sage_gemm_kernel<<<(tiles_u + 3) / 4, 128, 0, stream>>>(
        agg_u, deg_u, xu_cur, xu,
        wl_hi + off_u, wl_lo + off_u, bl_u, wr_hi + off_u, wr_lo + off_u, U, relu);
  }

  classifier_kernel<<<(L * 32 + B - 1) / B, B, 0, stream>>>(
      xu, xm, label_src, label_dst, out, L);
}
